// PatchTST_33930241638821
// MI455X (gfx1250) — compile-verified, matrix operand order checked by reading, not run on hardware
//
#include <hip/hip_runtime.h>

// ---------------- CDNA5 WMMA types ----------------
typedef unsigned short u16;
typedef unsigned int   u32;
typedef __attribute__((ext_vector_type(16))) __bf16 v16bf;
typedef __attribute__((ext_vector_type(8)))  float  v8f;

union Frag {
  v16bf v;
  u32   u[8];
  uint4 q[2];
};

__device__ __forceinline__ u16 f2bf(float f) {
  union { float f; u32 u; } x; x.f = f;
  return (u16)((x.u + 0x7fffu + ((x.u >> 16) & 1u)) >> 16);   // RNE truncate f32->bf16
}

// ---------------- Model dims ----------------
#define Bb    16
#define Lb    1024
#define MCH   21
#define Np    64
#define Pp    16
#define Dd    768
#define Hh    8
#define Ee    96
#define F4d   3072
#define NLAY  3
#define FREQn 33
#define A2n   32
#define PREDn 96
#define ROWS  (Bb*MCH*Np)      /* 21504 */
#define BM    (Bb*MCH)         /* 336   */
#define EPSc  1e-5f

// =====================================================================
// f32 -> bf16 bulk convert (weights once per launch; tiny vs 23.3 TB/s)
// =====================================================================
__global__ void f32_to_bf16(const float* __restrict__ s, u16* __restrict__ d, size_t n)
{
  size_t i = (size_t)blockIdx.x * blockDim.x + threadIdx.x;
  size_t stride = (size_t)gridDim.x * blockDim.x;
  for (; i < n; i += stride) d[i] = f2bf(s[i]);
}

// =====================================================================
// Generic GEMM:  O[M,N] = act( A[M,K] * W[N,K]^T + bias (+resid) )
// A and W are bf16; WMMA 16x16x32 bf16, f32 accumulate.
// Wave -> 16x64 C tile, 8 waves/block (4M x 2N) -> 64x128 block tile.
// Inner loop is pure global_load_b128 + v_wmma (no packing VALU).
// =====================================================================
template<bool OUTBF16, bool RELU, bool RESID, bool DUALOUT>
__global__ __launch_bounds__(256)
void gemm_bf16_wmma(const u16* __restrict__ A, int lda,
                    const u16* __restrict__ W,        // N x K row-major, bf16
                    const float* __restrict__ bias,   // N
                    const float* __restrict__ resid, int ldr,
                    void* __restrict__ Ov, int ldc,   // f32 or bf16 per OUTBF16
                    u16* __restrict__ O2,             // bf16 copy when DUALOUT
                    int M, int N, int K)
{
  const int tid  = threadIdx.x;
  const int wave = tid >> 5, lane = tid & 31;
  const int lrow = lane & 15, hi = lane >> 4;
  const int wm = wave & 3, wn = wave >> 2;
  const int row0 = blockIdx.x * 64 + wm * 16;
  const int col0 = blockIdx.y * 128 + wn * 64;
  if (row0 >= M || col0 >= N) return;       // whole-wave uniform exit; EXEC stays full

  v8f acc[4];
  for (int t = 0; t < 4; ++t)
    for (int i = 0; i < 8; ++i) acc[t][i] = 0.0f;

  const int arow = row0 + lrow;
  int bcol[4];
  for (int t = 0; t < 4; ++t) {
    int c = col0 + t * 16 + lrow;
    bcol[t] = (c < N) ? c : (N - 1);        // clamp; stores guarded later
  }

  const int k0 = hi * 8, k1 = 16 + hi * 8;  // ISA A/B fragment K layout: two b128 groups
  for (int kc = 0; kc < K; kc += 32) {
    Frag a;
    {
      const u16* p = A + (size_t)arow * lda + kc;
      a.q[0] = (kc + k0 < K) ? *(const uint4*)(p + k0) : make_uint4(0,0,0,0);
      a.q[1] = (kc + k1 < K) ? *(const uint4*)(p + k1) : make_uint4(0,0,0,0);
    }
    if (kc + 32 < K)   // hint next W chunk (global_prefetch_b8)
      __builtin_prefetch(W + (size_t)bcol[0] * K + kc + 32, 0, 0);

    for (int t = 0; t < 4; ++t) {
      Frag b;
      const u16* p = W + (size_t)bcol[t] * K + kc;
      b.q[0] = (kc + k0 < K) ? *(const uint4*)(p + k0) : make_uint4(0,0,0,0);
      b.q[1] = (kc + k1 < K) ? *(const uint4*)(p + k1) : make_uint4(0,0,0,0);
      acc[t] = __builtin_amdgcn_wmma_f32_16x16x32_bf16(
                 false, a.v, false, b.v, (short)0, acc[t], false, false);
    }
  }

  // ---- store ----
  for (int t = 0; t < 4; ++t) {
    int n = col0 + t * 16 + lrow;
    if (n >= N) continue;
    float bs = bias[n];
    for (int i = 0; i < 8; ++i) {
      int m = row0 + i + hi * 8;
      float v = acc[t][i] + bs;
      if (RESID) v += resid[(size_t)m * ldr + n];
      if (RELU)  v = fmaxf(v, 0.0f);
      if (OUTBF16) ((u16*)Ov)[(size_t)m * ldc + n] = f2bf(v);
      else         ((float*)Ov)[(size_t)m * ldc + n] = v;
      if (DUALOUT) O2[(size_t)m * ldc + n] = f2bf(v);
    }
  }
}

// =====================================================================
// Fused attention, one (bm,h) per block, 4 waves, wave -> 16 rows.
// scores = Q K^T/sqrt(E) + bias ; softmax ; O = attn V   (all WMMA bf16)
// =====================================================================
__global__ __launch_bounds__(128)
void attn_wmma(const u16* __restrict__ Q, const u16* __restrict__ Kb,
               const u16* __restrict__ Vb, const float* __restrict__ bias,
               u16* __restrict__ O)
{
  const int bh = blockIdx.x, bm = bh >> 3, h = bh & 7;
  const int tid = threadIdx.x;
  const int wave = tid >> 5, lane = tid & 31, lrow = lane & 15, hi = lane >> 4;
  const int l0 = wave * 16;
  const size_t base = (size_t)bm * Np * Dd + (size_t)h * Ee;

  __shared__ __align__(16) u16 attn_s[64][80];   // row stride 160B (16B aligned)
  __shared__ __align__(16) u16 vt[96][80];       // V^T: [e][s]

  for (int idx = tid; idx < Ee * Np; idx += 128) {
    int e = idx % Ee, s = idx / Ee;
    vt[e][s] = Vb[base + (size_t)s * Dd + e];
  }

  // ---- scores ----
  v8f sc[4];
  for (int t = 0; t < 4; ++t) for (int i = 0; i < 8; ++i) sc[t][i] = 0.0f;
  for (int kc = 0; kc < Ee; kc += 32) {
    Frag a;
    const u16* qp = Q + base + (size_t)(l0 + lrow) * Dd + kc;
    a.q[0] = *(const uint4*)(qp + hi * 8);
    a.q[1] = *(const uint4*)(qp + 16 + hi * 8);
    for (int t = 0; t < 4; ++t) {
      Frag b;
      const u16* kp = Kb + base + (size_t)(t * 16 + lrow) * Dd + kc;
      b.q[0] = *(const uint4*)(kp + hi * 8);
      b.q[1] = *(const uint4*)(kp + 16 + hi * 8);
      sc[t] = __builtin_amdgcn_wmma_f32_16x16x32_bf16(
                false, a.v, false, b.v, (short)0, sc[t], false, false);
    }
  }
  const float scale = 0.10206207261596575f;      // 1/sqrt(96)
  for (int t = 0; t < 4; ++t)
    for (int i = 0; i < 8; ++i) {
      int l = l0 + i + hi * 8, s = t * 16 + lrow;
      sc[t][i] = sc[t][i] * scale + bias[((size_t)l * Np + s) * Hh + h];
    }

  // ---- softmax (rows live in 16-lane halves; xor<16 stays in-half) ----
  for (int i = 0; i < 8; ++i) {
    float mx = -1e30f;
    for (int t = 0; t < 4; ++t) mx = fmaxf(mx, sc[t][i]);
    for (int off = 1; off < 16; off <<= 1) mx = fmaxf(mx, __shfl_xor(mx, off, 32));
    float sum = 0.0f;
    for (int t = 0; t < 4; ++t) { float e = __expf(sc[t][i] - mx); sc[t][i] = e; sum += e; }
    for (int off = 1; off < 16; off <<= 1) sum += __shfl_xor(sum, off, 32);
    float inv = 1.0f / sum;
    for (int t = 0; t < 4; ++t) sc[t][i] *= inv;
  }
  for (int t = 0; t < 4; ++t)
    for (int i = 0; i < 8; ++i) {
      int l = l0 + i + hi * 8, s = t * 16 + lrow;
      attn_s[l][s] = f2bf(sc[t][i]);
    }
  __syncthreads();

  // ---- O = attn @ V ----
  v8f oc[6];
  for (int t = 0; t < 6; ++t) for (int i = 0; i < 8; ++i) oc[t][i] = 0.0f;
  for (int kc = 0; kc < Np; kc += 32) {
    Frag a;
    const u16* ap = &attn_s[l0 + lrow][kc];
    a.q[0] = *(const uint4*)(ap + hi * 8);
    a.q[1] = *(const uint4*)(ap + 16 + hi * 8);
    for (int t = 0; t < 6; ++t) {
      Frag b;
      const u16* vp = &vt[t * 16 + lrow][kc];
      b.q[0] = *(const uint4*)(vp + hi * 8);
      b.q[1] = *(const uint4*)(vp + 16 + hi * 8);
      oc[t] = __builtin_amdgcn_wmma_f32_16x16x32_bf16(
                false, a.v, false, b.v, (short)0, oc[t], false, false);
    }
  }
  for (int t = 0; t < 6; ++t)
    for (int i = 0; i < 8; ++i) {
      int l = l0 + i + hi * 8, e = t * 16 + lrow;
      O[base + (size_t)l * Dd + e] = f2bf(oc[t][i]);
    }
}

// =====================================================================
// RevIN + patching: per (b,m) normalize over L; emit f32 + bf16 patches
// =====================================================================
__global__ __launch_bounds__(256)
void revin_patch(const float* __restrict__ x, const float* __restrict__ rw,
                 const float* __restrict__ rb, float* __restrict__ patches,
                 u16* __restrict__ patches_bf,
                 float* __restrict__ meanb, float* __restrict__ stdb)
{
  const int b = blockIdx.x / MCH, m = blockIdx.x % MCH;
  const int tid = threadIdx.x;
  float vals[4], s = 0.0f, sq = 0.0f;
  for (int j = 0; j < 4; ++j) {
    int l = tid + j * 256;
    float v = x[((size_t)b * Lb + l) * MCH + m];
    vals[j] = v; s += v; sq += v * v;
  }
  for (int off = 16; off; off >>= 1) { s += __shfl_xor(s, off, 32); sq += __shfl_xor(sq, off, 32); }
  __shared__ float sh[8], shq[8];
  __shared__ float mean_s, inv_s;
  if ((tid & 31) == 0) { sh[tid >> 5] = s; shq[tid >> 5] = sq; }
  __syncthreads();
  if (tid == 0) {
    float S = 0, Qq = 0;
    for (int w = 0; w < 8; ++w) { S += sh[w]; Qq += shq[w]; }
    float mean = S / (float)Lb;
    float var  = Qq / (float)Lb - mean * mean;
    float sd   = sqrtf(var + EPSc);
    meanb[blockIdx.x] = mean; stdb[blockIdx.x] = sd;
    mean_s = mean; inv_s = 1.0f / sd;
  }
  __syncthreads();
  const float g = rw[m], bb = rb[m];
  for (int j = 0; j < 4; ++j) {
    int l = tid + j * 256;
    float v = (vals[j] - mean_s) * inv_s * g + bb;
    patches[(size_t)blockIdx.x * Lb + l]    = v;
    patches_bf[(size_t)blockIdx.x * Lb + l] = f2bf(v);
  }
}

// =====================================================================
// Spectral block part 1: instance-norm across the M(=21) channel axis
// =====================================================================
__global__ void spect_norm(const float* __restrict__ patches,
                           const float* __restrict__ g, const float* __restrict__ b,
                           float* __restrict__ pn)
{
  int idx = blockIdx.x * blockDim.x + threadIdx.x;     // over B*N*P = B*1024
  if (idx >= Bb * 1024) return;
  int bb = idx >> 10, lp = idx & 1023;
  float v[MCH], s = 0.0f, sq = 0.0f;
  for (int m = 0; m < MCH; ++m) {
    float t = patches[((size_t)(bb * MCH + m)) * 1024 + lp];
    v[m] = t; s += t; sq += t * t;
  }
  float mu = s / (float)MCH, var = sq / (float)MCH - mu * mu;
  float inv = rsqrtf(var + EPSc);
  for (int m = 0; m < MCH; ++m)
    pn[((size_t)(bb * MCH + m)) * 1024 + lp] = (v[m] - mu) * inv * g[m] + b[m];
}

// =====================================================================
// Spectral block part 2: rFFT(64) -> proj(33->32) -> irFFT(64), per (b,m)
// result written directly as bf16 (only consumed by the adapter GEMM)
// =====================================================================
__global__ __launch_bounds__(256)
void spect_fft(const float* __restrict__ pn, const float* __restrict__ wr,
               const float* __restrict__ wi, u16* __restrict__ res)
{
  __shared__ float z[Pp][Np];
  __shared__ float re[Pp][FREQn], im[Pp][FREQn];
  __shared__ float zr[Pp][A2n], zi[Pp][A2n];
  const size_t base = (size_t)blockIdx.x * 1024;
  const int tid = threadIdx.x;
  const float W0 = 6.283185307179586f / 64.0f;

  for (int j = 0; j < 4; ++j) { int idx = tid + j * 256; z[idx & 15][idx >> 4] = pn[base + idx]; }
  __syncthreads();

  for (int t = tid; t < Pp * FREQn; t += 256) {
    int p = t / FREQn, f = t % FREQn;
    float sr = 0.0f, si = 0.0f;
    for (int n = 0; n < Np; ++n) {
      float ang = W0 * (float)(f * n);
      sr += z[p][n] * __cosf(ang);
      si -= z[p][n] * __sinf(ang);
    }
    re[p][f] = sr; im[p][f] = si;
  }
  __syncthreads();

  for (int t = tid; t < Pp * A2n; t += 256) {
    int p = t >> 5, d = t & 31;
    float ar = 0.0f, ai = 0.0f;
    for (int f = 0; f < FREQn; ++f) { ar += re[p][f] * wr[f * A2n + d]; ai += im[p][f] * wi[f * A2n + d]; }
    zr[p][d] = ar; zi[p][d] = ai;
  }
  __syncthreads();

  for (int j = 0; j < 4; ++j) {
    int idx = tid + j * 256;
    int p = idx & 15, a = idx >> 4;
    float v = zr[p][0];
    for (int d = 1; d < A2n; ++d) {
      float ang = W0 * (float)(d * a);
      v += 2.0f * (zr[p][d] * __cosf(ang) - zi[p][d] * __sinf(ang));
    }
    res[base + idx] = f2bf(v * (1.0f / 64.0f));
  }
}

// =====================================================================
// BatchNorm over rows per feature d; stats then apply (+opt bf16 copy)
// =====================================================================
__global__ __launch_bounds__(256)
void bn_stats(const float* __restrict__ y, int rows, int ld,
              float* __restrict__ mu, float* __restrict__ var)
{
  const int d = blockIdx.x;
  float s = 0.0f, sq = 0.0f;
  for (int r = threadIdx.x; r < rows; r += 256) {
    float v = y[(size_t)r * ld + d];
    s += v; sq += v * v;
  }
  for (int off = 16; off; off >>= 1) { s += __shfl_xor(s, off, 32); sq += __shfl_xor(sq, off, 32); }
  __shared__ float sh[8], shq[8];
  if ((threadIdx.x & 31) == 0) { sh[threadIdx.x >> 5] = s; shq[threadIdx.x >> 5] = sq; }
  __syncthreads();
  if (threadIdx.x == 0) {
    float S = 0, Qq = 0;
    for (int w = 0; w < 8; ++w) { S += sh[w]; Qq += shq[w]; }
    float m = S / (float)rows;
    mu[d] = m; var[d] = Qq / (float)rows - m * m;
  }
}

__global__ void bn_apply(float* __restrict__ y, const float* __restrict__ mu,
                         const float* __restrict__ var, const float* __restrict__ g,
                         const float* __restrict__ b, size_t total, int ld,
                         u16* __restrict__ obf)
{
  size_t i = (size_t)blockIdx.x * blockDim.x + threadIdx.x;
  size_t stride = (size_t)gridDim.x * blockDim.x;
  for (; i < total; i += stride) {
    int d = (int)(i % (size_t)ld);
    float v = (y[i] - mu[d]) * rsqrtf(var[d] + EPSc) * g[d] + b[d];
    y[i] = v;
    if (obf) obf[i] = f2bf(v);
  }
}

// =====================================================================
// Output: (B,M,PRED) -> (B,PRED,M) with inverse RevIN
// =====================================================================
__global__ void final_out(const float* __restrict__ t336, const float* __restrict__ rw,
                          const float* __restrict__ rb, const float* __restrict__ meanb,
                          const float* __restrict__ stdb, float* __restrict__ out)
{
  int idx = blockIdx.x * blockDim.x + threadIdx.x;
  if (idx >= Bb * PREDn * MCH) return;
  int m = idx % MCH, rest = idx / MCH, p = rest % PREDn, b = rest / PREDn;
  int bm = b * MCH + m;
  float v = t336[(size_t)bm * PREDn + p];
  out[idx] = (v - rb[m]) / (rw[m] + 1e-10f) * stdb[bm] + meanb[bm];
}

// =====================================================================
// Host side
// =====================================================================
template<bool OUTBF16, bool RELU, bool RESID, bool DUALOUT>
static void launch_gemm(const u16* A, int lda, const u16* W, const float* bias,
                        const float* resid, int ldr, void* O, int ldc, u16* O2,
                        int M, int N, int K, hipStream_t s)
{
  dim3 grid((M + 63) / 64, (N + 127) / 128);
  gemm_bf16_wmma<OUTBF16, RELU, RESID, DUALOUT><<<grid, 256, 0, s>>>(
      A, lda, W, bias, resid, ldr, O, ldc, O2, M, N, K);
}

static inline size_t aln(size_t x) { return (x + 255) & ~(size_t)255; }

extern "C" void kernel_launch(void* const* d_in, const int* in_sizes, int n_in,
                              void* d_out, int out_size, void* d_ws, size_t ws_size,
                              hipStream_t stream)
{
  (void)in_sizes; (void)n_in; (void)out_size; (void)ws_size;
  const float* x         = (const float*)d_in[0];
  const float* attn_bias = (const float*)d_in[1];
  const float* revin_w   = (const float*)d_in[2];
  const float* revin_b   = (const float*)d_in[3];
  const float* in_w      = (const float*)d_in[4];
  const float* in_b      = (const float*)d_in[5];
  const float* ln1_g     = (const float*)d_in[6];
  const float* ln1_b     = (const float*)d_in[7];
  const float* spect_wr  = (const float*)d_in[8];
  const float* spect_wi  = (const float*)d_in[9];
  const float* Wq = (const float*)d_in[10]; const float* bq = (const float*)d_in[11];
  const float* Wk = (const float*)d_in[12]; const float* bk = (const float*)d_in[13];
  const float* Wv = (const float*)d_in[14]; const float* bv = (const float*)d_in[15];
  const float* Wo = (const float*)d_in[16]; const float* bo = (const float*)d_in[17];
  const float* c1w = (const float*)d_in[18]; const float* c1b = (const float*)d_in[19];
  const float* c2w = (const float*)d_in[20]; const float* c2b = (const float*)d_in[21];
  const float* bn1g = (const float*)d_in[22]; const float* bn1b = (const float*)d_in[23];
  const float* bn2g = (const float*)d_in[24]; const float* bn2b = (const float*)d_in[25];
  const float* normg = (const float*)d_in[26]; const float* normb = (const float*)d_in[27];
  const float* out_w = (const float*)d_in[28]; const float* out_b = (const float*)d_in[29];

  // ---- workspace carve ----
  char* ws = (char*)d_ws;
  size_t off = 0;
  auto carve = [&](size_t bytes) { char* p = ws + off; off += aln(bytes); return (void*)p; };
  float* patches = (float*)carve(sizeof(float) * BM * 1024);
  u16*   patches_bf = (u16*)carve(sizeof(u16) * BM * 1024);
  float* pnbuf   = (float*)carve(sizeof(float) * BM * 1024);
  u16*   sres_bf = (u16*)  carve(sizeof(u16) * BM * 1024);
  float* meanb   = (float*)carve(sizeof(float) * BM);
  float* stdb    = (float*)carve(sizeof(float) * BM);
  float* mu      = (float*)carve(sizeof(float) * Dd);
  float* var     = (float*)carve(sizeof(float) * Dd);
  float* bufH    = (float*)carve(sizeof(float) * (size_t)ROWS * Dd);
  float* bufY    = (float*)carve(sizeof(float) * (size_t)ROWS * Dd);
  float* bufT    = (float*)carve(sizeof(float) * (size_t)ROWS * Dd);
  u16*   ybf     = (u16*)  carve(sizeof(u16)   * (size_t)ROWS * Dd);
  u16*   tbf     = (u16*)  carve(sizeof(u16)   * (size_t)ROWS * Dd);
  u16*   hbf     = (u16*)  carve(sizeof(u16)   * (size_t)ROWS * Dd);
  u16*   qb      = (u16*)  carve(sizeof(u16)   * (size_t)ROWS * Dd);
  u16*   kb      = (u16*)  carve(sizeof(u16)   * (size_t)ROWS * Dd);
  u16*   vb      = (u16*)  carve(sizeof(u16)   * (size_t)ROWS * Dd);
  u16*   ob      = (u16*)  carve(sizeof(u16)   * (size_t)ROWS * Dd);
  const int CH = ROWS / 8;                                // 2688-row FFN chunks
  u16*   fbuf    = (u16*)  carve(sizeof(u16)   * (size_t)CH * F4d);
  float* headb   = (float*)carve(sizeof(float) * BM * PREDn);
  // bf16 weights (converted once per launch)
  u16* inw_bf  = (u16*)carve(sizeof(u16) * (size_t)Dd * Pp);
  u16* wq_bf   = (u16*)carve(sizeof(u16) * (size_t)NLAY * Dd * Dd);
  u16* wk_bf   = (u16*)carve(sizeof(u16) * (size_t)NLAY * Dd * Dd);
  u16* wv_bf   = (u16*)carve(sizeof(u16) * (size_t)NLAY * Dd * Dd);
  u16* wo_bf   = (u16*)carve(sizeof(u16) * (size_t)NLAY * Dd * Dd);
  u16* c1_bf   = (u16*)carve(sizeof(u16) * (size_t)NLAY * F4d * Dd);
  u16* c2_bf   = (u16*)carve(sizeof(u16) * (size_t)NLAY * Dd * F4d);
  u16* outw_bf = (u16*)carve(sizeof(u16) * (size_t)PREDn * Np * Dd);

  const size_t totalHD = (size_t)ROWS * Dd;
  auto cvt = [&](const float* s, u16* d, size_t n) {
    size_t blocks = (n + 255) / 256; if (blocks > 2048) blocks = 2048;
    f32_to_bf16<<<(unsigned)blocks, 256, 0, stream>>>(s, d, n);
  };

  // 0) one-time weight conversion (streamed once; HBM-bound, ~100 MB)
  cvt(in_w,  inw_bf,  (size_t)Dd * Pp);
  cvt(Wq,    wq_bf,   (size_t)NLAY * Dd * Dd);
  cvt(Wk,    wk_bf,   (size_t)NLAY * Dd * Dd);
  cvt(Wv,    wv_bf,   (size_t)NLAY * Dd * Dd);
  cvt(Wo,    wo_bf,   (size_t)NLAY * Dd * Dd);
  cvt(c1w,   c1_bf,   (size_t)NLAY * F4d * Dd);
  cvt(c2w,   c2_bf,   (size_t)NLAY * Dd * F4d);
  cvt(out_w, outw_bf, (size_t)PREDn * Np * Dd);

  // 1) RevIN + patching
  revin_patch<<<BM, 256, 0, stream>>>(x, revin_w, revin_b, patches, patches_bf, meanb, stdb);

  // 2) h = patches @ in_w.T + in_b    (K=16, zero-padded to 32 inside)
  launch_gemm<false,false,false,false>(patches_bf, Pp, inw_bf, in_b, nullptr, 0,
                                       bufH, Dd, nullptr, ROWS, Dd, Pp, stream);

  for (int i = 0; i < NLAY; ++i) {
    // spectral adapter
    spect_norm<<<(Bb*1024 + 255)/256, 256, 0, stream>>>(patches, ln1_g + i*MCH, ln1_b + i*MCH, pnbuf);
    spect_fft<<<BM, 256, 0, stream>>>(pnbuf, spect_wr + i*FREQn*A2n, spect_wi + i*FREQn*A2n, sres_bf);
    // y = h + ad   (dual out: f32 residual carrier + bf16 for QKV GEMMs)
    launch_gemm<false,false,true,true>(sres_bf, Pp, inw_bf, in_b, bufH, Dd,
                                       bufY, Dd, ybf, ROWS, Dd, Pp, stream);
    // Q, K, V (bf16 out for the attention kernel)
    launch_gemm<true,false,false,false>(ybf, Dd, wq_bf + (size_t)i*Dd*Dd, bq + i*Dd,
                                        nullptr, 0, qb, Dd, nullptr, ROWS, Dd, Dd, stream);
    launch_gemm<true,false,false,false>(ybf, Dd, wk_bf + (size_t)i*Dd*Dd, bk + i*Dd,
                                        nullptr, 0, kb, Dd, nullptr, ROWS, Dd, Dd, stream);
    launch_gemm<true,false,false,false>(ybf, Dd, wv_bf + (size_t)i*Dd*Dd, bv + i*Dd,
                                        nullptr, 0, vb, Dd, nullptr, ROWS, Dd, Dd, stream);
    // fused attention
    attn_wmma<<<BM * Hh, 128, 0, stream>>>(qb, kb, vb, attn_bias, ob);
    // y = y + attnO @ Wo.T + bo
    launch_gemm<false,false,true,false>(ob, Dd, wo_bf + (size_t)i*Dd*Dd, bo + i*Dd,
                                        bufY, Dd, bufT, Dd, nullptr, ROWS, Dd, Dd, stream);
    // batchnorm 1 (bf16 copy feeds FFN1)
    bn_stats<<<Dd, 256, 0, stream>>>(bufT, ROWS, Dd, mu, var);
    bn_apply<<<4096, 256, 0, stream>>>(bufT, mu, var, bn1g + i*Dd, bn1b + i*Dd, totalHD, Dd, tbf);
    // FFN, chunked
    for (int c = 0; c < 8; ++c) {
      u16*   ycb = tbf  + (size_t)c * CH * Dd;
      float* ycf = bufT + (size_t)c * CH * Dd;
      float* hc  = bufH + (size_t)c * CH * Dd;
      launch_gemm<true,true,false,false>(ycb, Dd, c1_bf + (size_t)i*F4d*Dd, c1b + i*F4d,
                                         nullptr, 0, fbuf, F4d, nullptr, CH, F4d, Dd, stream);
      launch_gemm<false,false,true,false>(fbuf, F4d, c2_bf + (size_t)i*Dd*F4d, c2b + i*Dd,
                                          ycf, Dd, hc, Dd, nullptr, CH, Dd, F4d, stream);
    }
    // batchnorm 2 -> new h (f32 only; used as residual / stats next)
    bn_stats<<<Dd, 256, 0, stream>>>(bufH, ROWS, Dd, mu, var);
    bn_apply<<<4096, 256, 0, stream>>>(bufH, mu, var, bn2g + i*Dd, bn2b + i*Dd, totalHD, Dd, nullptr);
  }

  // final norm (bf16 copy feeds the head GEMM)
  bn_stats<<<Dd, 256, 0, stream>>>(bufH, ROWS, Dd, mu, var);
  bn_apply<<<4096, 256, 0, stream>>>(bufH, mu, var, normg, normb, totalHD, Dd, hbf);

  // head: (336, 49152) @ out_w.T -> (336, 96)
  launch_gemm<false,false,false,false>(hbf, Np * Dd, outw_bf, out_b, nullptr, 0,
                                       headb, PREDn, nullptr, BM, PREDn, Np * Dd, stream);

  // inverse RevIN + transpose to (B, PRED, M)
  final_out<<<(Bb*PREDn*MCH + 255)/256, 256, 0, stream>>>(
      headb, revin_w, revin_b, meanb, stdb, (float*)d_out);
}
